// HLFAtt_50122268344642
// MI455X (gfx1250) — matrix operand order repaired and checked by reading, not compiled, not run on hardware
//
#include <hip/hip_runtime.h>

// ---- model constants ----
#define DIMC 256
#define HD   32
#define LD_  128   // low-freq dim (4 heads x 32)
#define HDIM 128   // high-freq dim
#define BB   32
#define HGT  64
#define WID  64
#define HW   4096
#define NKV  256   // pooled tokens per batch
#define SCALE_ 0.17677669529663687f  // 32^-0.5

typedef __attribute__((ext_vector_type(16))) __bf16 v16bf;
typedef __attribute__((ext_vector_type(8)))  __bf16 v8bf;
typedef __attribute__((ext_vector_type(8)))  float  v8f;

__device__ __forceinline__ v8f wmma_bf16(v16bf a, v16bf b, v8f c) {
  return __builtin_amdgcn_wmma_f32_16x16x32_bf16(false, a, false, b, (short)0, c,
                                                 false, false);
}

// A-tile 16x32 (MxK), row-major source, base = &M[0][k0], stride ld (elems).
// Lane l: m=l&15, half=l>>4; elems 0..15 = K[half*16+0..15].
// Two 16-byte vector loads per lane (16B alignment guaranteed by callers).
__device__ __forceinline__ v16bf ldA(const __bf16* p, int ld) {
  int lane = threadIdx.x & 31;
  int m = lane & 15, hh = lane >> 4;
  const __bf16* r = p + m * ld + hh * 16;
  v8bf lo = *(const v8bf*)(r);
  v8bf hi = *(const v8bf*)(r + 8);
  return __builtin_shufflevector(lo, hi, 0, 1, 2, 3, 4, 5, 6, 7,
                                 8, 9, 10, 11, 12, 13, 14, 15);
}

// Pre-packed B fragment: tile = 32 lanes x 16 contiguous bf16 (512 elems, 1KB).
// One 32-byte vector load per lane.
__device__ __forceinline__ v16bf ldBp(const __bf16* tile) {
  return *(const v16bf*)(tile + (threadIdx.x & 31) * 16);
}

// B-tile where B[k][n] = src[n][k] (B = srcT), src row-major stride ld (elems),
// base = &src[n0][k0]. One contiguous 32-byte load per lane (32B aligned callers).
__device__ __forceinline__ v16bf ldBt(const __bf16* p, int ld) {
  int lane = threadIdx.x & 31;
  int n = lane & 15, hh = lane >> 4;
  return *(const v16bf*)(p + n * ld + hh * 16);
}

// B-tile with only K rows 0..15 valid (rows 16..31 zero); for the K=16 P@V GEMM.
__device__ __forceinline__ v16bf ldB_k16(const __bf16* p, int ld) {
  int lane = threadIdx.x & 31;
  int n = lane & 15, hh = lane >> 4;
  v16bf b;
  if (hh == 0) {
    const __bf16* c = p + n;
#pragma unroll
    for (int e = 0; e < 16; ++e) b[e] = c[e * ld];
  } else {
#pragma unroll
    for (int e = 0; e < 16; ++e) b[e] = (__bf16)0.0f;
  }
  return b;
}

// Store D (16x16 f32 acc) as bf16 row-major with scale. Lane l: n=l&15,
// rows m = (l>>4)*8 + r  (ISA C/D layout).
__device__ __forceinline__ void stD_bf16(__bf16* p, int ld, v8f d, float s) {
  int lane = threadIdx.x & 31;
  int n = lane & 15, hh = lane >> 4;
#pragma unroll
  for (int r = 0; r < 8; ++r) p[(hh * 8 + r) * ld + n] = (__bf16)(d[r] * s);
}

// ---------------- weight pack: f32 [K][N] -> WMMA B-fragment layout ----------------
// dst[((kt*(N/16)+nt)*32 + lane)*16 + e] = W[kt*32 + (lane>>4)*16 + e][nt*16 + (lane&15)]
__global__ void k_packB(const float* __restrict__ src, __bf16* __restrict__ dst,
                        int K, int N) {
  int i = blockIdx.x * 256 + threadIdx.x;
  if (i >= K * N) return;
  int e = i & 15, lane = (i >> 4) & 31, t = i >> 9;
  int ntiles = N >> 4;
  int kt = t / ntiles, nt = t % ntiles;
  int n = lane & 15, hh = lane >> 4;
  int k = kt * 32 + hh * 16 + e;
  dst[i] = (__bf16)src[k * N + nt * 16 + n];
}

// Same, but B[k][n] = src[n][k] with src row-major [N][K] (conv_w is (out,in)).
__global__ void k_packBT(const float* __restrict__ src, __bf16* __restrict__ dst,
                         int K, int N) {
  int i = blockIdx.x * 256 + threadIdx.x;
  if (i >= K * N) return;
  int e = i & 15, lane = (i >> 4) & 31, t = i >> 9;
  int ntiles = N >> 4;
  int kt = t / ntiles, nt = t % ntiles;
  int n = lane & 15, hh = lane >> 4;
  int k = kt * 32 + hh * 16 + e;
  dst[i] = (__bf16)src[(nt * 16 + n) * K + k];
}

// Repack V half of kv[b][key][256] (cols 128..255) into per-batch B fragments.
__global__ void k_packV(const __bf16* __restrict__ kv, __bf16* __restrict__ vp) {
  int i = blockIdx.x * 256 + threadIdx.x;          // over BB * 256*128
  if (i >= BB * NKV * LD_) return;
  int b = i >> 15, j = i & 32767;                  // 256*128 = 32768 per batch
  int e = j & 15, lane = (j >> 4) & 31, t = j >> 9; // t = kt*8 + nt
  int kt = t >> 3, nt = t & 7;
  int n = lane & 15, hh = lane >> 4;
  int k = kt * 32 + hh * 16 + e;                   // key index
  vp[i] = kv[((size_t)b * NKV + k) * DIMC + 128 + nt * 16 + n];
}

// ---------------- avg pool 4x4 -> bf16 token-major [b][pt][c] ----------------
__global__ void k_pool(const float* __restrict__ x, __bf16* __restrict__ pooled) {
  int i = blockIdx.x * 256 + threadIdx.x;
  if (i >= BB * NKV * DIMC) return;
  int c  = i & 255;
  int pt = (i >> 8) & 255;
  int b  = i >> 16;
  int py = pt >> 4, px = pt & 15;
  const float* xb = x + (size_t)b * DIMC * HW + (size_t)c * HW;
  float s = 0.f;
#pragma unroll
  for (int dy = 0; dy < 4; ++dy)
#pragma unroll
    for (int dx = 0; dx < 4; ++dx)
      s += xb[(py * 4 + dy) * WID + px * 4 + dx];
  pooled[i] = (__bf16)(s * 0.0625f);
}

// ---------------- kv = pooled @ lkv -> kvb[b][key][256] (k:0..127, v:128..255) ----
__global__ void __launch_bounds__(128) k_lkv(const __bf16* __restrict__ pooled,
                                             const __bf16* __restrict__ lkvp,
                                             __bf16* __restrict__ kvout) {
  int b = blockIdx.x >> 4, mt = blockIdx.x & 15;
  int wv = threadIdx.x >> 5;
  const __bf16* A0 = pooled + ((size_t)b * NKV + mt * 16) * DIMC;
  __bf16* D0 = kvout + ((size_t)b * NKV + mt * 16) * DIMC;
  for (int nt = wv; nt < 16; nt += 4) {   // 256 output cols
    v8f acc = {};
#pragma unroll
    for (int k0 = 0; k0 < DIMC; k0 += 32)
      acc = wmma_bf16(ldA(A0 + k0, DIMC), ldBp(lkvp + ((k0 >> 5) * 16 + nt) * 512), acc);
    stD_bf16(D0 + nt * 16, DIMC, acc, 1.0f);
  }
}

// ---------------- high-frequency windowed attention ----------------
__global__ void __launch_bounds__(128) k_hf(const float* __restrict__ x,
                                            const __bf16* __restrict__ hqkvp,
                                            const __bf16* __restrict__ hprojp,
                                            const float* __restrict__ hproj_b,
                                            __bf16* __restrict__ hf_out) {
  __shared__ alignas(32) __bf16 xt[16][DIMC];     // 8 KB
  __shared__ alignas(32) __bf16 qkv[16][384];     // 12 KB  q:0..127 k:128..255 v:256..383
  __shared__ alignas(32) __bf16 P[4][16][32];     // 4 KB
  __shared__ alignas(32) __bf16 O[16][HDIM];      // 4 KB
  int b = blockIdx.x >> 8, g = blockIdx.x & 255;
  int gy = g >> 4, gx = g & 15;
  int h = threadIdx.x >> 5;           // wave == head
  int lane = threadIdx.x & 31;
  const float* xb = x + (size_t)b * DIMC * HW;

  for (int f = threadIdx.x; f < 16 * DIMC; f += 128) {
    int c = f >> 4, m = f & 15;
    xt[m][c] = (__bf16)xb[(size_t)c * HW + (gy * 4 + (m >> 2)) * WID + gx * 4 + (m & 3)];
  }
  __syncthreads();

  for (int nt = h; nt < 24; nt += 4) {       // qkv: 16x256 @ 256x384
    v8f acc = {};
#pragma unroll
    for (int k0 = 0; k0 < DIMC; k0 += 32)
      acc = wmma_bf16(ldA(&xt[0][0] + k0, DIMC), ldBp(hqkvp + ((k0 >> 5) * 24 + nt) * 512), acc);
    stD_bf16(&qkv[0][0] + nt * 16, 384, acc, 1.0f);
  }
  __syncthreads();

  {                                           // S = Q Kt * scale  (16x16, K=32)
    v8f z = {};
    v8f s = wmma_bf16(ldA(&qkv[0][0] + h * 32, 384),
                      ldBt(&qkv[0][0] + 128 + h * 32, 384), z);
    stD_bf16(&P[h][0][0], 32, s, SCALE_);
  }
  __syncthreads();

  if (lane < 16) {                            // row softmax + zero-pad cols 16..31
    __bf16* row = &P[h][lane][0];
    float mx = -3.4e38f;
    for (int j = 0; j < 16; ++j) mx = fmaxf(mx, (float)row[j]);
    float sum = 0.f;
    for (int j = 0; j < 16; ++j) { float e = __expf((float)row[j] - mx); sum += e; row[j] = (__bf16)e; }
    float inv = 1.f / sum;
    for (int j = 0; j < 16; ++j) row[j] = (__bf16)((float)row[j] * inv);
    for (int j = 16; j < 32; ++j) row[j] = (__bf16)0.f;
  }
  __syncthreads();

#pragma unroll
  for (int nt = 0; nt < 2; ++nt) {            // O_h = P @ V_h  (K=16 padded)
    v8f z = {};
    v8f o = wmma_bf16(ldA(&P[h][0][0], 32),
                      ldB_k16(&qkv[0][0] + 256 + h * 32 + nt * 16, 384), z);
    stD_bf16(&O[0][0] + h * 32 + nt * 16, HDIM, o, 1.0f);
  }
  __syncthreads();

  for (int nt = h; nt < 8; nt += 4) {         // hproj + bias -> hf_out (raster tokens)
    v8f acc = {};
#pragma unroll
    for (int k0 = 0; k0 < HDIM; k0 += 32)
      acc = wmma_bf16(ldA(&O[0][0] + k0, HDIM), ldBp(hprojp + ((k0 >> 5) * 8 + nt) * 512), acc);
    int n = lane & 15, hl = lane >> 4;
    float bv = hproj_b[nt * 16 + n];
#pragma unroll
    for (int r = 0; r < 8; ++r) {
      int m = hl * 8 + r;
      int tok = (gy * 4 + (m >> 2)) * WID + gx * 4 + (m & 3);
      hf_out[((size_t)b * HW + tok) * HDIM + nt * 16 + n] = (__bf16)(acc[r] + bv);
    }
  }
}

// ---------------- low-frequency pooled global attention ----------------
__global__ void __launch_bounds__(128) k_lf(const float* __restrict__ x,
                                            const __bf16* __restrict__ lqp,
                                            const __bf16* __restrict__ kvb,
                                            const __bf16* __restrict__ vp,
                                            const __bf16* __restrict__ lprojp,
                                            const float* __restrict__ lproj_b,
                                            __bf16* __restrict__ lf_out) {
  __shared__ alignas(32) __bf16 xt[16][DIMC];     // 8 KB
  __shared__ alignas(32) __bf16 qld[16][LD_];     // 4 KB
  __shared__ alignas(32) __bf16 S[4][16][NKV];    // 32 KB  scores -> probs in place
  __shared__ alignas(32) __bf16 O[16][LD_];       // 4 KB
  int b = blockIdx.x >> 8;
  int t0 = (blockIdx.x & 255) * 16;   // 16 consecutive raster tokens
  int h = threadIdx.x >> 5, lane = threadIdx.x & 31;
  const float* xb = x + (size_t)b * DIMC * HW;

  for (int f = threadIdx.x; f < 16 * DIMC; f += 128) {
    int c = f >> 4, m = f & 15;
    xt[m][c] = (__bf16)xb[(size_t)c * HW + t0 + m];
  }
  __syncthreads();

#pragma unroll
  for (int i = 0; i < 2; ++i) {               // q = xt @ lq (cols of own head)
    int nt = h * 2 + i;
    v8f acc = {};
#pragma unroll
    for (int k0 = 0; k0 < DIMC; k0 += 32)
      acc = wmma_bf16(ldA(&xt[0][0] + k0, DIMC), ldBp(lqp + ((k0 >> 5) * 8 + nt) * 512), acc);
    stD_bf16(&qld[0][0] + nt * 16, LD_, acc, 1.0f);
  }
  __syncthreads();

  const __bf16* kvbase = kvb + (size_t)b * NKV * DIMC;
  for (int nk = 0; nk < 16; ++nk) {           // S = q @ Kt over 256 keys
    v8f z = {};
    v8f s = wmma_bf16(ldA(&qld[0][0] + h * 32, LD_),
                      ldBt(kvbase + (size_t)(nk * 16) * DIMC + h * 32, DIMC), z);
    stD_bf16(&S[h][0][0] + nk * 16, NKV, s, SCALE_);
  }
  __syncthreads();

  if (lane < 16) {                            // softmax over 256 keys, in place
    __bf16* row = &S[h][lane][0];
    float mx = -3.4e38f;
    for (int j = 0; j < NKV; ++j) mx = fmaxf(mx, (float)row[j]);
    float sum = 0.f;
    for (int j = 0; j < NKV; ++j) { float e = __expf((float)row[j] - mx); sum += e; row[j] = (__bf16)e; }
    float inv = 1.f / sum;
    for (int j = 0; j < NKV; ++j) row[j] = (__bf16)((float)row[j] * inv);
  }
  __syncthreads();

  const __bf16* vpb = vp + (size_t)b * (NKV * LD_);
#pragma unroll
  for (int nt = 0; nt < 2; ++nt) {            // O_h = P @ V_h  (K=256)
    v8f acc = {};
#pragma unroll
    for (int k0 = 0; k0 < NKV; k0 += 32)
      acc = wmma_bf16(ldA(&S[h][0][0] + k0, NKV),
                      ldBp(vpb + ((k0 >> 5) * 8 + (h * 2 + nt)) * 512), acc);
    stD_bf16(&O[0][0] + h * 32 + nt * 16, LD_, acc, 1.0f);
  }
  __syncthreads();

  __bf16* dst = lf_out + ((size_t)b * HW + t0) * LD_;
  for (int nt = h; nt < 8; nt += 4) {         // lproj + bias
    v8f acc = {};
#pragma unroll
    for (int k0 = 0; k0 < LD_; k0 += 32)
      acc = wmma_bf16(ldA(&O[0][0] + k0, LD_), ldBp(lprojp + ((k0 >> 5) * 8 + nt) * 512), acc);
    int n = lane & 15, hl = lane >> 4;
    float bv = lproj_b[nt * 16 + n];
#pragma unroll
    for (int r = 0; r < 8; ++r)
      dst[(hl * 8 + r) * LD_ + nt * 16 + n] = (__bf16)(acc[r] + bv);
  }
}

// ---------------- concat + 1x1 conv + LeakyReLU -> NCHW f32 ----------------
__global__ void __launch_bounds__(128) k_conv(const __bf16* __restrict__ hf_b,
                                              const __bf16* __restrict__ lf_b,
                                              const __bf16* __restrict__ convp,
                                              const float* __restrict__ conv_b,
                                              float* __restrict__ out) {
  __shared__ alignas(32) __bf16 ct[16][DIMC];
  int b = blockIdx.x >> 8;
  int t0 = (blockIdx.x & 255) * 16;
  int h = threadIdx.x >> 5, lane = threadIdx.x & 31;
  for (int f = threadIdx.x; f < 16 * DIMC; f += 128) {
    int m = f >> 8, c = f & 255;
    size_t tok = (size_t)b * HW + t0 + m;
    ct[m][c] = (c < HDIM) ? hf_b[tok * HDIM + c] : lf_b[tok * LD_ + (c - HDIM)];
  }
  __syncthreads();
  float* ob = out + (size_t)b * DIMC * HW;
  for (int nt = h; nt < 16; nt += 4) {        // 256 output channels
    v8f acc = {};
#pragma unroll
    for (int k0 = 0; k0 < DIMC; k0 += 32)
      acc = wmma_bf16(ldA(&ct[0][0] + k0, DIMC), ldBp(convp + ((k0 >> 5) * 16 + nt) * 512), acc);
    int n = lane & 15, hl = lane >> 4;
    float bv = conv_b[nt * 16 + n];
#pragma unroll
    for (int r = 0; r < 8; ++r) {
      float v = acc[r] + bv;
      v = (v > 0.f) ? v : 0.01f * v;
      ob[(size_t)(nt * 16 + n) * HW + t0 + hl * 8 + r] = v;
    }
  }
}

// ---------------- host launch ----------------
extern "C" void kernel_launch(void* const* d_in, const int* in_sizes, int n_in,
                              void* d_out, int out_size, void* d_ws, size_t ws_size,
                              hipStream_t stream) {
  (void)in_sizes; (void)n_in; (void)out_size; (void)ws_size;
  const float* x       = (const float*)d_in[0];
  const float* lq_w    = (const float*)d_in[1];
  const float* lkv_w   = (const float*)d_in[2];
  const float* lproj_w = (const float*)d_in[3];
  const float* lproj_b = (const float*)d_in[4];
  const float* hqkv_w  = (const float*)d_in[5];
  const float* hproj_w = (const float*)d_in[6];
  const float* hproj_b = (const float*)d_in[7];
  const float* conv_w  = (const float*)d_in[8];
  const float* conv_b  = (const float*)d_in[9];
  float* out = (float*)d_out;
  char* ws = (char*)d_ws;

  // workspace layout (bytes; all offsets 1KB-multiples so fragment loads stay 32B aligned)
  const size_t OFF_HQKV  = 0;                       // 256*384 bf16 (packed)
  const size_t OFF_LQ    = OFF_HQKV + 98304 * 2;    // 256*128
  const size_t OFF_LKV   = OFF_LQ   + 32768 * 2;    // 256*256
  const size_t OFF_LPROJ = OFF_LKV  + 65536 * 2;    // 128*128
  const size_t OFF_HPROJ = OFF_LPROJ + 16384 * 2;   // 128*128
  const size_t OFF_CONVT = OFF_HPROJ + 16384 * 2;   // 256*256 (transposed+packed)
  const size_t OFF_POOL  = OFF_CONVT + 65536 * 2;   // 32*256*256
  const size_t OFF_KV    = OFF_POOL + (size_t)BB * NKV * DIMC * 2;
  const size_t OFF_VP    = OFF_KV   + (size_t)BB * NKV * DIMC * 2;  // packed V frags
  const size_t OFF_HF    = OFF_VP   + (size_t)BB * NKV * LD_ * 2;
  const size_t OFF_LF    = OFF_HF   + (size_t)BB * HW * HDIM * 2;

  __bf16* hqkv_p  = (__bf16*)(ws + OFF_HQKV);
  __bf16* lq_p    = (__bf16*)(ws + OFF_LQ);
  __bf16* lkv_p   = (__bf16*)(ws + OFF_LKV);
  __bf16* lproj_p = (__bf16*)(ws + OFF_LPROJ);
  __bf16* hproj_p = (__bf16*)(ws + OFF_HPROJ);
  __bf16* conv_p  = (__bf16*)(ws + OFF_CONVT);
  __bf16* pool_bf = (__bf16*)(ws + OFF_POOL);
  __bf16* kv_bf   = (__bf16*)(ws + OFF_KV);
  __bf16* v_p     = (__bf16*)(ws + OFF_VP);
  __bf16* hf_bf   = (__bf16*)(ws + OFF_HF);
  __bf16* lf_bf   = (__bf16*)(ws + OFF_LF);

  k_packB <<<98304 / 256, 256, 0, stream>>>(hqkv_w, hqkv_p, 256, 384);
  k_packB <<<32768 / 256, 256, 0, stream>>>(lq_w, lq_p, 256, 128);
  k_packB <<<65536 / 256, 256, 0, stream>>>(lkv_w, lkv_p, 256, 256);
  k_packB <<<16384 / 256, 256, 0, stream>>>(lproj_w, lproj_p, 128, 128);
  k_packB <<<16384 / 256, 256, 0, stream>>>(hproj_w, hproj_p, 128, 128);
  k_packBT<<<65536 / 256, 256, 0, stream>>>(conv_w, conv_p, 256, 256);

  k_pool<<<(BB * NKV * DIMC) / 256, 256, 0, stream>>>(x, pool_bf);
  k_lkv<<<BB * 16, 128, 0, stream>>>(pool_bf, lkv_p, kv_bf);
  k_packV<<<(BB * NKV * LD_) / 256, 256, 0, stream>>>(kv_bf, v_p);

  k_hf<<<BB * 256, 128, 0, stream>>>(x, hqkv_p, hproj_p, hproj_b, hf_bf);
  k_lf<<<BB * 256, 128, 0, stream>>>(x, lq_p, kv_bf, v_p, lproj_p, lproj_b, lf_bf);

  k_conv<<<BB * 256, 128, 0, stream>>>(hf_bf, lf_bf, conv_p, conv_b, out);
}
